// TransformerLayer_49426483642951
// MI455X (gfx1250) — matrix operand order repaired and checked by reading, not compiled, not run on hardware
//
#include <hip/hip_runtime.h>
#include <hip/hip_bf16.h>
#include <cstdint>

// ---- problem constants -----------------------------------------------------
#define BB 4
#define SS 2048
#define DD 1024
#define HH 16
#define HD 64
#define D3 3072
#define ROWS (BB * SS)          // 8192
#define LN_EPS 1e-5f

typedef __attribute__((ext_vector_type(16))) __bf16 v16bf;
typedef __attribute__((ext_vector_type(8)))  float  v8f;
typedef __attribute__((ext_vector_type(4)))  int    v4i;

typedef __attribute__((address_space(1))) v4i* gv4i_p;   // global
typedef __attribute__((address_space(3))) v4i* lv4i_p;   // LDS

// ---- gfx1250 async global->LDS support (probe via __has_builtin) -----------
#define HAS_ASYNC_LDS 0
#define HAS_WAIT_ASYNC 0
#if defined(__has_builtin)
#  if __has_builtin(__builtin_amdgcn_global_load_async_to_lds_b128)
#    undef HAS_ASYNC_LDS
#    define HAS_ASYNC_LDS 1
#  endif
#  if __has_builtin(__builtin_amdgcn_s_wait_asynccnt)
#    undef HAS_WAIT_ASYNC
#    define HAS_WAIT_ASYNC 1
#  endif
#endif

__device__ __forceinline__ void async_copy16(const uint16_t* g, uint16_t* l) {
#if HAS_ASYNC_LDS
    __builtin_amdgcn_global_load_async_to_lds_b128((gv4i_p)g, (lv4i_p)l, 0, 0);
#else
    *(uint4*)l = *(const uint4*)g;          // sync fallback (reg -> ds_store)
#endif
}

__device__ __forceinline__ void wait_async_copies() {
#if HAS_ASYNC_LDS
#  if HAS_WAIT_ASYNC
    __builtin_amdgcn_s_wait_asynccnt(0);
#  else
    asm volatile("s_wait_asynccnt 0x0" ::: "memory");
#  endif
#endif
}

__device__ __forceinline__ uint16_t f2bf(float f) {
    uint32_t u = __builtin_bit_cast(uint32_t, f);
    u += 0x7FFFu + ((u >> 16) & 1u);          // round-to-nearest-even
    return (uint16_t)(u >> 16);
}

__device__ __forceinline__ v8f wmma_bf16(v16bf a, v16bf b, v8f c) {
    return __builtin_amdgcn_wmma_f32_16x16x32_bf16(
        /*neg_a=*/false, a, /*neg_b=*/false, b,
        /*c_mod=*/(short)0, c, /*reuse_a=*/false, /*reuse_b=*/false);
}

union Frag { v16bf v; uint4 u[2]; };

// ---- fp32 -> bf16 cast -----------------------------------------------------
__global__ void k_f32_to_bf16(const float* __restrict__ s, uint16_t* __restrict__ d, int n) {
    for (int i = blockIdx.x * blockDim.x + threadIdx.x; i < n; i += gridDim.x * blockDim.x)
        d[i] = f2bf(s[i]);
}

// ---- layernorm (per row of 1024), bf16 output ------------------------------
__global__ __launch_bounds__(256) void k_layernorm(const float* __restrict__ x,
                                                   const float* __restrict__ g,
                                                   const float* __restrict__ bta,
                                                   uint16_t* __restrict__ y) {
    const int row = blockIdx.x;
    const int tid = threadIdx.x;
    const float* xr = x + (size_t)row * DD;

    float v[4], s = 0.f, ss = 0.f;
#pragma unroll
    for (int i = 0; i < 4; ++i) {
        v[i] = xr[tid + i * 256];
        s += v[i];
        ss += v[i] * v[i];
    }
#pragma unroll
    for (int d2 = 16; d2; d2 >>= 1) {
        s  += __shfl_xor(s,  d2, 32);
        ss += __shfl_xor(ss, d2, 32);
    }
    __shared__ float sm[8], sm2[8];
    const int w = tid >> 5, l = tid & 31;
    if (l == 0) { sm[w] = s; sm2[w] = ss; }
    __syncthreads();
    float ts = 0.f, tss = 0.f;
#pragma unroll
    for (int j = 0; j < 8; ++j) { ts += sm[j]; tss += sm2[j]; }
    const float mean = ts * (1.f / DD);
    const float var  = tss * (1.f / DD) - mean * mean;
    const float rstd = rsqrtf(var + LN_EPS);
#pragma unroll
    for (int i = 0; i < 4; ++i) {
        const int c = tid + i * 256;
        y[(size_t)row * DD + c] = f2bf((v[i] - mean) * rstd * g[c] + bta[c]);
    }
}

// ---- bf16 WMMA GEMM, LDS double-buffered via async global->LDS -------------
// C = A(MxK) @ W(NxK)^T.  Block tile 128x128, K-step 32, 8 waves (2m x 4n),
// wave tile 64x32 = 4x2 WMMA tiles.
// mode 0: write bf16 C           -> outBF
// mode 1: outF = resid + relu(C)
// mode 2: outF = resid + relu(C + bias[n])
__global__ __launch_bounds__(256) void k_gemm_bf16(const uint16_t* __restrict__ A,
                                                   const uint16_t* __restrict__ W,
                                                   int M, int N, int K, int mode,
                                                   const float* __restrict__ resid,
                                                   const float* __restrict__ bias,
                                                   float* __restrict__ outF,
                                                   uint16_t* __restrict__ outBF) {
    const int tid  = threadIdx.x;
    const int lane = tid & 31;
    const int wave = tid >> 5;
    const int lm = lane & 15, lh = lane >> 4;
    const int wn = wave & 3;          // 4 waves across N (32 cols each)
    const int wm = wave >> 2;         // 2 waves across M (64 rows each)
    const int mRow0 = blockIdx.y * 128;
    const int nCol0 = blockIdx.x * 128;

    // [buf][slab(0=A,1=B)][128 rows x 32 k]
    __shared__ alignas(16) uint16_t smem[2][2][128 * 32];

    const v8f z = {0.f, 0.f, 0.f, 0.f, 0.f, 0.f, 0.f, 0.f};
    v8f acc[4][2];
#pragma unroll
    for (int i = 0; i < 4; ++i)
#pragma unroll
        for (int j = 0; j < 2; ++j) acc[i][j] = z;

    // block-cooperative slab copy: 512 x 16B chunks per slab, 2 per thread
    auto copySlabs = [&](int k0, int buf) {
#pragma unroll
        for (int t = 0; t < 2; ++t) {
            const int c    = tid + t * 256;          // [0, 512)
            const int row  = c >> 2;                 // [0, 128)
            const int offE = (c & 3) * 8;            // element offset in k
            async_copy16(A + (size_t)(mRow0 + row) * K + k0 + offE,
                         &smem[buf][0][row * 32 + offE]);
            async_copy16(W + (size_t)(nCol0 + row) * K + k0 + offE,
                         &smem[buf][1][row * 32 + offE]);
        }
    };

    copySlabs(0, 0);
    wait_async_copies();
    __syncthreads();

    for (int k0 = 0; k0 < K; k0 += 32) {
        const int cur = (k0 >> 5) & 1;
        const bool more = (k0 + 32) < K;
        if (more) copySlabs(k0 + 32, cur ^ 1);       // overlap with compute

        const uint16_t* sA = smem[cur][0];
        const uint16_t* sB = smem[cur][1];

        Frag bF[2];
#pragma unroll
        for (int j = 0; j < 2; ++j) {
            const int col = wn * 32 + j * 16 + lm;   // B: k chunks at lh*8, lh*8+16
            bF[j].u[0] = *(const uint4*)&sB[col * 32 + lh * 8];
            bF[j].u[1] = *(const uint4*)&sB[col * 32 + lh * 8 + 16];
        }
#pragma unroll
        for (int i = 0; i < 4; ++i) {
            const int row = wm * 64 + i * 16 + lm;   // A: k chunks at lh*8, lh*8+16
            Frag aF;
            aF.u[0] = *(const uint4*)&sA[row * 32 + lh * 8];
            aF.u[1] = *(const uint4*)&sA[row * 32 + lh * 8 + 16];
#pragma unroll
            for (int j = 0; j < 2; ++j)
                acc[i][j] = wmma_bf16(aF.v, bF[j].v, acc[i][j]);
        }

        if (more) wait_async_copies();
        __syncthreads();
    }

#pragma unroll
    for (int i = 0; i < 4; ++i)
#pragma unroll
        for (int j = 0; j < 2; ++j)
#pragma unroll
            for (int r = 0; r < 8; ++r) {
                const int m = mRow0 + wm * 64 + i * 16 + r + lh * 8;  // C layout
                const int n = nCol0 + wn * 32 + j * 16 + lm;
                const size_t idx = (size_t)m * N + n;
                const float c = acc[i][j][r];
                if (mode == 0)      outBF[idx] = f2bf(c);
                else if (mode == 1) outF[idx] = resid[idx] + fmaxf(c, 0.f);
                else                outF[idx] = resid[idx] + fmaxf(c + bias[n], 0.f);
            }
}

// ---- build V^T [B*H][HD][S] from qkv (V = slot 1 of (q,v,k)) ---------------
__global__ void k_transpose_v(const uint16_t* __restrict__ qkv, uint16_t* __restrict__ vT, int total) {
    for (int i = blockIdx.x * blockDim.x + threadIdx.x; i < total; i += gridDim.x * blockDim.x) {
        const int s  = i & (SS - 1);
        const int hd = (i >> 11) & (HD - 1);
        const int bh = i >> 17;
        const int b = bh >> 4, h = bh & (HH - 1);
        vT[i] = qkv[((size_t)b * SS + s) * D3 + DD + h * HD + hd];  // +DD: V slot
    }
}

// ---- causal flash attention: one wave per (b,h,16-row block) --------------
__global__ __launch_bounds__(32) void k_attention(const uint16_t* __restrict__ qkv,
                                                  const uint16_t* __restrict__ vT,
                                                  uint16_t* __restrict__ attn) {
    const int lane = threadIdx.x & 31;
    const int lm = lane & 15, lh = lane >> 4;
    const int m0 = blockIdx.x * 16;
    const int bh = blockIdx.y;
    const int b = bh >> 4, h = bh & (HH - 1);

    const size_t rowQ = ((size_t)b * SS + m0 + lm) * D3 + h * HD;
    Frag qA[2];
#pragma unroll
    for (int kk = 0; kk < 2; ++kk) {
        const uint16_t* p = qkv + rowQ + kk * 32 + lh * 8;
        qA[kk].u[0] = *(const uint4*)(p);
        qA[kk].u[1] = *(const uint4*)(p + 16);
    }

    const v8f z = {0.f, 0.f, 0.f, 0.f, 0.f, 0.f, 0.f, 0.f};
    v8f o[4];
    float rm[8], rl[8];
#pragma unroll
    for (int n2 = 0; n2 < 4; ++n2) o[n2] = z;
#pragma unroll
    for (int r = 0; r < 8; ++r) { rm[r] = -1e30f; rl[r] = 0.f; }

    __shared__ alignas(16) uint16_t pT[16 * 32];
    const uint16_t* vBase = vT + (size_t)bh * HD * SS;

    for (int t0 = 0; t0 < m0 + 16; t0 += 32) {
        v8f sc[2];
#pragma unroll
        for (int tt = 0; tt < 2; ++tt) {
            const uint16_t* kp = qkv + ((size_t)b * SS + t0 + tt * 16 + lm) * D3
                                 + 2 * DD + h * HD + lh * 8;
            Frag kB;
            kB.u[0] = *(const uint4*)(kp);
            kB.u[1] = *(const uint4*)(kp + 16);
            v8f t = wmma_bf16(qA[0].v, kB.v, z);
            kB.u[0] = *(const uint4*)(kp + 32);
            kB.u[1] = *(const uint4*)(kp + 48);
            sc[tt] = wmma_bf16(qA[1].v, kB.v, t);
        }

#pragma unroll
        for (int r = 0; r < 8; ++r) {
            const int m = m0 + r + lh * 8;
            float s0 = (t0 + lm      <= m) ? sc[0][r] : -1e30f;
            float s1 = (t0 + 16 + lm <= m) ? sc[1][r] : -1e30f;
            float mx = fmaxf(s0, s1);
#pragma unroll
            for (int d2 = 8; d2; d2 >>= 1) mx = fmaxf(mx, __shfl_xor(mx, d2, 32));
            const float nm = fmaxf(rm[r], mx);
            const float al = __expf(rm[r] - nm);
            const float p0 = __expf(s0 - nm);
            const float p1 = __expf(s1 - nm);
            float ps = p0 + p1;
#pragma unroll
            for (int d2 = 8; d2; d2 >>= 1) ps += __shfl_xor(ps, d2, 32);
            rl[r] = rl[r] * al + ps;
            rm[r] = nm;
#pragma unroll
            for (int n2 = 0; n2 < 4; ++n2) o[n2][r] *= al;
            pT[(r + lh * 8) * 32 + lm]      = f2bf(p0);
            pT[(r + lh * 8) * 32 + 16 + lm] = f2bf(p1);
        }
        __syncthreads();

        Frag pA;
        pA.u[0] = *(const uint4*)(pT + lm * 32 + lh * 8);
        pA.u[1] = *(const uint4*)(pT + lm * 32 + lh * 8 + 16);
#pragma unroll
        for (int n2 = 0; n2 < 4; ++n2) {
            const uint16_t* vp = vBase + (size_t)(n2 * 16 + lm) * SS + t0 + lh * 8;
            Frag vB;
            vB.u[0] = *(const uint4*)(vp);
            vB.u[1] = *(const uint4*)(vp + 16);
            o[n2] = wmma_bf16(pA.v, vB.v, o[n2]);
        }
        __syncthreads();
    }

#pragma unroll
    for (int r = 0; r < 8; ++r) {
        const float inv = 1.0f / rl[r];
        const size_t base = ((size_t)b * SS + m0 + r + lh * 8) * DD + h * HD;
#pragma unroll
        for (int n2 = 0; n2 < 4; ++n2)
            attn[base + n2 * 16 + lm] = f2bf(o[n2][r] * inv);
    }
}

// ---- launcher --------------------------------------------------------------
extern "C" void kernel_launch(void* const* d_in, const int* in_sizes, int n_in,
                              void* d_out, int out_size, void* d_ws, size_t ws_size,
                              hipStream_t stream) {
    const float* h     = (const float*)d_in[0];
    const float* w_qkv = (const float*)d_in[1];
    const float* w_o   = (const float*)d_in[2];
    const float* w_fc  = (const float*)d_in[3];
    const float* b_fc  = (const float*)d_in[4];
    const float* ln1_g = (const float*)d_in[5];
    const float* ln1_b = (const float*)d_in[6];
    const float* ln2_g = (const float*)d_in[7];
    const float* ln2_b = (const float*)d_in[8];
    float* out = (float*)d_out;

    uint16_t* w_qkv_bf = (uint16_t*)d_ws;                    // 3*D*D
    uint16_t* w_o_bf   = w_qkv_bf + (size_t)3 * DD * DD;     // D*D
    uint16_t* w_fc_bf  = w_o_bf   + (size_t)DD * DD;         // D*D
    uint16_t* y_bf     = w_fc_bf  + (size_t)DD * DD;         // ROWS*D
    uint16_t* qkv_bf   = y_bf     + (size_t)ROWS * DD;       // ROWS*3D
    uint16_t* vT_bf    = qkv_bf   + (size_t)ROWS * D3;       // B*H*HD*S
    uint16_t* attn_bf  = vT_bf    + (size_t)ROWS * DD;       // ROWS*D
    float*    h1       = (float*)(attn_bf + (size_t)ROWS * DD); // ROWS*D f32
    uint16_t* z_bf     = (uint16_t*)(h1 + (size_t)ROWS * DD);   // ROWS*D

    k_f32_to_bf16<<<4096, 256, 0, stream>>>(w_qkv, w_qkv_bf, 3 * DD * DD);
    k_f32_to_bf16<<<2048, 256, 0, stream>>>(w_o,   w_o_bf,   DD * DD);
    k_f32_to_bf16<<<2048, 256, 0, stream>>>(w_fc,  w_fc_bf,  DD * DD);

    k_layernorm<<<ROWS, 256, 0, stream>>>(h, ln1_g, ln1_b, y_bf);

    k_gemm_bf16<<<dim3(D3 / 128, ROWS / 128), 256, 0, stream>>>(
        y_bf, w_qkv_bf, ROWS, D3, DD, 0, nullptr, nullptr, nullptr, qkv_bf);

    k_transpose_v<<<8192, 256, 0, stream>>>(qkv_bf, vT_bf, ROWS * DD);

    k_attention<<<dim3(SS / 16, BB * HH), 32, 0, stream>>>(qkv_bf, vT_bf, attn_bf);

    k_gemm_bf16<<<dim3(DD / 128, ROWS / 128), 256, 0, stream>>>(
        attn_bf, w_o_bf, ROWS, DD, DD, 1, h, nullptr, h1, nullptr);

    k_layernorm<<<ROWS, 256, 0, stream>>>(h1, ln2_g, ln2_b, z_bf);

    k_gemm_bf16<<<dim3(DD / 128, ROWS / 128), 256, 0, stream>>>(
        z_bf, w_fc_bf, ROWS, DD, DD, 2, h1, b_fc, out, nullptr);
}
